// EncoderBlock_30494267802165
// MI455X (gfx1250) — matrix operand order repaired and checked by reading, not compiled, not run on hardware
//
#include <hip/hip_runtime.h>
#include <math.h>

// ---------------------------------------------------------------------------
// Transformer encoder block on gfx1250 (MI455X), bf16 WMMA everywhere.
// B=16 S=1024 D=768 H=12 DH=64 DFF=3072.  ~290 GFLOP/call -> compute bound
// on the matrix pipes; all GEMMs + attention use v_wmma_f32_16x16x32_bf16.
// Tile staging and A-fragment LDS reads are vectorized (global_load_b128 / ds_load_b128);
// each wave carries two 16-row strips to double WMMA issue per staged tile.
// ---------------------------------------------------------------------------

#define Bq 16
#define Sq 1024
#define Dq 768
#define Hq 12
#define DHq 64
#define DFFq 3072
#define Mq (Bq * Sq)          // 16384 rows of the activation stream

typedef __attribute__((ext_vector_type(16))) __bf16 v16bf;
typedef __attribute__((ext_vector_type(8)))  __bf16 v8bf;
typedef __attribute__((ext_vector_type(8)))  float  v8f;

// A fragment = two contiguous 8-elem (16B) runs per lane (ISA 7.12.2):
// lanes 0-15 : K in [0..7] and [16..23];  lanes 16-31: K in [8..15] and [24..31]
union AFrag { v16bf v; v8bf h[2]; };

__device__ __forceinline__ v8f wmma_bf16(v16bf a, v16bf b, v8f c) {
    return __builtin_amdgcn_wmma_f32_16x16x32_bf16(
        /*neg_a=*/false, a, /*neg_b=*/false, b,
        /*c_mod=*/(short)0, c, /*reuse_a=*/false, /*reuse_b=*/false);
}

// ---------------------------------------------------------------------------
// Weight prep: fuse Wq/Wk/Wv [H,D,DH] into bf16 [D, 3*768] (K-major for GEMM B)
// ---------------------------------------------------------------------------
__global__ void k_pack_wqkv(const float* __restrict__ wq,
                            const float* __restrict__ wk,
                            const float* __restrict__ wv,
                            __bf16* __restrict__ out) {
    int tid = blockIdx.x * blockDim.x + threadIdx.x;
    const int per = Dq * (Hq * DHq);          // 589824
    if (tid >= 3 * per) return;
    int j = tid / per;
    int r = tid - j * per;
    int d = r / (Hq * DHq);
    int n = r - d * (Hq * DHq);               // n = h*64 + k
    int h = n >> 6, k = n & 63;
    const float* src = (j == 0) ? wq : (j == 1) ? wk : wv;
    float v = src[(size_t)h * Dq * DHq + (size_t)d * DHq + k];
    out[(size_t)d * (3 * Dq) + j * Dq + n] = (__bf16)v;
}

__global__ void k_pack_bqkv(const float* __restrict__ bq,
                            const float* __restrict__ bk,
                            const float* __restrict__ bv,
                            float* __restrict__ out) {
    int tid = blockIdx.x * blockDim.x + threadIdx.x;
    if (tid >= 3 * Dq) return;
    int j = tid / Dq, n = tid - j * Dq;       // [H,DH] contiguous as h*DH+k
    const float* src = (j == 0) ? bq : (j == 1) ? bk : bv;
    out[tid] = src[n];
}

__global__ void k_cvt_bf16(const float* __restrict__ in,
                           __bf16* __restrict__ out, int n) {
    int tid = blockIdx.x * blockDim.x + threadIdx.x;
    if (tid < n) out[tid] = (__bf16)in[tid];
}

// ---------------------------------------------------------------------------
// LayerNorm over D=768: one block per row, 3 elems/thread, LDS reduction.
// ---------------------------------------------------------------------------
__global__ __launch_bounds__(256) void k_ln(const float* __restrict__ x,
                                            const float* __restrict__ g,
                                            const float* __restrict__ b,
                                            __bf16* __restrict__ out) {
    const int row = blockIdx.x, tid = threadIdx.x;
    const float* xr = x + (size_t)row * Dq;
    float v0 = xr[tid], v1 = xr[tid + 256], v2 = xr[tid + 512];
    __shared__ float rs[256], rss[256];
    rs[tid]  = v0 + v1 + v2;
    rss[tid] = v0 * v0 + v1 * v1 + v2 * v2;
    __syncthreads();
    for (int off = 128; off > 0; off >>= 1) {
        if (tid < off) { rs[tid] += rs[tid + off]; rss[tid] += rss[tid + off]; }
        __syncthreads();
    }
    float mu  = rs[0] * (1.0f / Dq);
    float var = rss[0] * (1.0f / Dq) - mu * mu;
    float inv = rsqrtf(var + 1e-5f);
    __bf16* o = out + (size_t)row * Dq;
    o[tid]       = (__bf16)((v0 - mu) * inv * g[tid]       + b[tid]);
    o[tid + 256] = (__bf16)((v1 - mu) * inv * g[tid + 256] + b[tid + 256]);
    o[tid + 512] = (__bf16)((v2 - mu) * inv * g[tid + 512] + b[tid + 512]);
}

// ---------------------------------------------------------------------------
// Tiled WMMA GEMM:  out[M,N] = A[M,K](bf16) * Bm[K,N](bf16) + bias [+epilogue]
// Block = 256 threads = 8 waves; block tile 256x64, K-step 32.
// Each wave: two 16-row strips, 8 WMMAs per K-step sharing 4 B fragments.
// EPI: 1 = bias, store bf16                 (QKV proj)
//      2 = bias + exact GELU, store bf16    (MLP up-proj)
//      3 = bias + residual(f32), store f32  (O proj / MLP down-proj)
// ---------------------------------------------------------------------------
template <int EPI>
__global__ __launch_bounds__(256, 2)
void k_gemm(const __bf16* __restrict__ A, const __bf16* __restrict__ Bm,
            const float* __restrict__ bias, const float* __restrict__ res,
            void* __restrict__ out, int M, int N, int K) {
    const int tid = threadIdx.x, lane = tid & 31, w = tid >> 5;
    const int hi8 = (lane & 16) >> 1;         // 0 or 8
    const int cb0 = blockIdx.x * 64;
    const int rb0 = blockIdx.y * 256;
    __shared__ alignas(32) __bf16 Asm[256 * 32];   // 16 KB
    __shared__ alignas(32) __bf16 Bsm[32 * 64];    //  4 KB

    v8f acc[2][4] = {};
    const int arow0 = (w * 32 + (lane & 15)) * 32; // strip 0 row base in LDS
    const int arow1 = arow0 + 16 * 32;             // strip 1

    for (int kt = 0; kt < K; kt += 32) {
        __syncthreads();
        // stage A tile 256x32: 1024 uint4 (16B) loads, 4 per thread
        #pragma unroll
        for (int i = tid; i < 1024; i += 256) {
            int r = i >> 2, c = i & 3;
            ((uint4*)Asm)[i] =
                *(const uint4*)(A + (size_t)(rb0 + r) * K + kt + c * 8);
        }
        // stage B tile 32x64: 256 uint4, 1 per thread
        {
            int r = tid >> 3, c = tid & 7;
            ((uint4*)Bsm)[tid] =
                *(const uint4*)(Bm + (size_t)(kt + r) * N + cb0 + c * 8);
        }
        if (kt + 32 < K) {
            __builtin_prefetch(A + (size_t)(rb0 + tid) * K + kt + 32, 0, 1);
            __builtin_prefetch(Bm + (size_t)(kt + 32 + (tid >> 3)) * N + cb0, 0, 1);
        }
        __syncthreads();

        AFrag a0, a1;                              // two strips, 16B LDS loads
        a0.h[0] = *(const v8bf*)(Asm + arow0 + hi8);
        a0.h[1] = *(const v8bf*)(Asm + arow0 + 16 + hi8);
        a1.h[0] = *(const v8bf*)(Asm + arow1 + hi8);
        a1.h[1] = *(const v8bf*)(Asm + arow1 + 16 + hi8);

        // B fragments: lane = column n, elems = K in table order, shared by both strips
        v16bf b0, b1, b2, b3;
        #pragma unroll
        for (int e = 0; e < 16; e++) {
            const __bf16* bp = Bsm + (hi8 + e + (e & 8)) * 64 + (lane & 15);
            b0[e] = bp[0]; b1[e] = bp[16]; b2[e] = bp[32]; b3[e] = bp[48];
        }
        acc[0][0] = wmma_bf16(a0.v, b0, acc[0][0]);
        acc[1][0] = wmma_bf16(a1.v, b0, acc[1][0]);
        acc[0][1] = wmma_bf16(a0.v, b1, acc[0][1]);
        acc[1][1] = wmma_bf16(a1.v, b1, acc[1][1]);
        acc[0][2] = wmma_bf16(a0.v, b2, acc[0][2]);
        acc[1][2] = wmma_bf16(a1.v, b2, acc[1][2]);
        acc[0][3] = wmma_bf16(a0.v, b3, acc[0][3]);
        acc[1][3] = wmma_bf16(a1.v, b3, acc[1][3]);
    }

    // epilogue: C layout -> row = r + hi8, col = chunk*16 + lane%16
    #pragma unroll
    for (int s = 0; s < 2; s++) {
        #pragma unroll
        for (int c = 0; c < 4; c++) {
            #pragma unroll
            for (int r = 0; r < 8; r++) {
                float v = acc[s][c][r];
                int row = rb0 + w * 32 + s * 16 + r + hi8;
                int col = cb0 + c * 16 + (lane & 15);
                v += bias[col];
                if (EPI == 2)
                    v = 0.5f * v * (1.0f + erff(v * 0.70710678118654752f));
                size_t o = (size_t)row * N + col;
                if (EPI == 3) ((float*)out)[o] = v + res[o];
                else          ((__bf16*)out)[o] = (__bf16)v;
            }
        }
    }
}

// ---------------------------------------------------------------------------
// Flash attention, transposed-score formulation, 32 queries per wave.
// grid (S/256, H, B), block = 8 waves; wave w owns Q rows [row0, row0+32)
// as two 16-query groups. Per 32-key step: S^T = K_tile * Q^T (K tile is the
// A operand -> contiguous LDS loads, shared by both query groups; Q^T
// B-fragments are loop-invariant registers with the 1/sqrt(DH) scale folded
// in). The S^T C-layout maps lane-for-lane onto the P A-fragment layout, so
// softmaxed probabilities feed the P*V WMMAs directly from registers.
// qkv layout: [B*S, 2304] bf16 with Q at +0, K at +768, V at +1536.
// ---------------------------------------------------------------------------
__global__ __launch_bounds__(256)
void k_attn(const __bf16* __restrict__ qkv, __bf16* __restrict__ ctx) {
    const int bb_ = blockIdx.z, hh = blockIdx.y;
    const int tid = threadIdx.x, lane = tid & 31, w = tid >> 5;
    const int q   = lane & 15;
    const int hi8 = (lane & 16) >> 1;           // 0 or 8
    const int row0 = blockIdx.x * 256 + w * 32;

    __shared__ alignas(32) __bf16 Kt[32 * 64];
    __shared__ alignas(32) __bf16 Vt[32 * 64];

    // Q^T B-fragments (loop-invariant): lane%16 = query, elem = dh (table order).
    // 1/sqrt(64) folded in (exact: power-of-two scale).
    v16bf bqf[2][2];                            // [query group][dh chunk]
    #pragma unroll
    for (int e = 0; e < 16; e++) {
        const __bf16* qp =
            qkv + (size_t)(bb_ * Sq + row0 + q) * (3 * Dq) + hh * DHq + hi8 + e + (e & 8);
        bqf[0][0][e] = (__bf16)(0.125f * (float)qp[0]);
        bqf[0][1][e] = (__bf16)(0.125f * (float)qp[32]);
        bqf[1][0][e] = (__bf16)(0.125f * (float)qp[16 * 3 * Dq]);
        bqf[1][1][e] = (__bf16)(0.125f * (float)qp[16 * 3 * Dq + 32]);
    }

    v8f acc[2][4] = {};
    float mcur[2] = {-1e30f, -1e30f}, lcur[2] = {0.0f, 0.0f};

    for (int t = 0; t < Sq; t += 32) {
        __syncthreads();
        // stage K/V tiles (32 keys x 64 dh): 256 uint4 each, 1 per thread
        {
            int r = tid >> 3, c = tid & 7;
            size_t eoff = (size_t)(bb_ * Sq + t + r) * (3 * Dq) + Dq +
                          hh * DHq + c * 8;
            ((uint4*)Kt)[tid] = *(const uint4*)(qkv + eoff);
            ((uint4*)Vt)[tid] = *(const uint4*)(qkv + eoff + Dq);
        }
        __syncthreads();

        // S^T = K * Q^T : per group, two 16(keys) x 16(queries) tiles, K=64.
        // K A-fragments are loaded once and shared by both query groups.
        v8f s0[2] = {}, s1[2] = {};
        {
            AFrag ak;
            const int b00 = q * 64 + hi8;       // keys t..t+15
            ak.h[0] = *(const v8bf*)(Kt + b00);
            ak.h[1] = *(const v8bf*)(Kt + b00 + 16);
            s0[0] = wmma_bf16(ak.v, bqf[0][0], s0[0]);
            s0[1] = wmma_bf16(ak.v, bqf[1][0], s0[1]);
            ak.h[0] = *(const v8bf*)(Kt + b00 + 32);
            ak.h[1] = *(const v8bf*)(Kt + b00 + 48);
            s0[0] = wmma_bf16(ak.v, bqf[0][1], s0[0]);
            s0[1] = wmma_bf16(ak.v, bqf[1][1], s0[1]);
            const int b10 = (16 + q) * 64 + hi8; // keys t+16..t+31
            ak.h[0] = *(const v8bf*)(Kt + b10);
            ak.h[1] = *(const v8bf*)(Kt + b10 + 16);
            s1[0] = wmma_bf16(ak.v, bqf[0][0], s1[0]);
            s1[1] = wmma_bf16(ak.v, bqf[1][0], s1[1]);
            ak.h[0] = *(const v8bf*)(Kt + b10 + 32);
            ak.h[1] = *(const v8bf*)(Kt + b10 + 48);
            s1[0] = wmma_bf16(ak.v, bqf[0][1], s1[0]);
            s1[1] = wmma_bf16(ak.v, bqf[1][1], s1[1]);
        }

        // V B-fragments (shared by both groups): lane = dh col, elems = keys in table order
        v16bf bv0, bv1, bv2, bv3;
        #pragma unroll
        for (int e = 0; e < 16; e++) {
            const __bf16* vp = Vt + (hi8 + e + (e & 8)) * 64 + q;
            bv0[e] = vp[0]; bv1[e] = vp[16]; bv2[e] = vp[32]; bv3[e] = vp[48];
        }

        // online softmax per group: per-lane = one query, 16 in-lane keys.
        #pragma unroll
        for (int g = 0; g < 2; g++) {
            float mloc = -1e30f;
            #pragma unroll
            for (int r = 0; r < 8; r++)
                mloc = fmaxf(mloc, fmaxf(s0[g][r], s1[g][r]));
            mloc = fmaxf(mloc, __shfl_xor(mloc, 16));
            float mnew  = fmaxf(mcur[g], mloc);
            float alpha = __expf(mcur[g] - mnew);
            AFrag ap;                           // P A-fragment, built in regs
            float sl = 0.0f;
            #pragma unroll
            for (int r = 0; r < 8; r++) {
                float e0 = __expf(s0[g][r] - mnew);
                float e1 = __expf(s1[g][r] - mnew);
                sl += e0 + e1;
                ap.h[0][r] = (__bf16)e0;        // keys hi8 + r
                ap.h[1][r] = (__bf16)e1;        // keys 16 + hi8 + r
            }
            sl += __shfl_xor(sl, 16);
            lcur[g] = lcur[g] * alpha + sl;
            mcur[g] = mnew;

            // rescale accumulators: acc row r holds query (r + hi8); alpha
            // for query qq lives in lane qq -> broadcast with 8 shfls.
            float av[8];
            #pragma unroll
            for (int r = 0; r < 8; r++) av[r] = __shfl(alpha, r + hi8);
            #pragma unroll
            for (int r = 0; r < 8; r++) {
                acc[g][0][r] *= av[r]; acc[g][1][r] *= av[r];
                acc[g][2][r] *= av[r]; acc[g][3][r] *= av[r];
            }

            acc[g][0] = wmma_bf16(ap.v, bv0, acc[g][0]);
            acc[g][1] = wmma_bf16(ap.v, bv1, acc[g][1]);
            acc[g][2] = wmma_bf16(ap.v, bv2, acc[g][2]);
            acc[g][3] = wmma_bf16(ap.v, bv3, acc[g][3]);
        }
    }

    // normalize + store ctx (concat-heads layout [B*S, 768])
    #pragma unroll
    for (int g = 0; g < 2; g++) {
        float inv = 1.0f / lcur[g];
        float iv[8];
        #pragma unroll
        for (int r = 0; r < 8; r++) iv[r] = __shfl(inv, r + hi8);
        #pragma unroll
        for (int r = 0; r < 8; r++) {
            int row = row0 + g * 16 + r + hi8;
            size_t o = (size_t)(bb_ * Sq + row) * Dq + hh * DHq + q;
            ctx[o]      = (__bf16)(acc[g][0][r] * iv[r]);
            ctx[o + 16] = (__bf16)(acc[g][1][r] * iv[r]);
            ctx[o + 32] = (__bf16)(acc[g][2][r] * iv[r]);
            ctx[o + 48] = (__bf16)(acc[g][3][r] * iv[r]);
        }
    }
}

// ---------------------------------------------------------------------------
// Host launcher
// ---------------------------------------------------------------------------
static inline size_t alignup(size_t x) { return (x + 255) & ~(size_t)255; }

extern "C" void kernel_launch(void* const* d_in, const int* in_sizes, int n_in,
                              void* d_out, int out_size, void* d_ws, size_t ws_size,
                              hipStream_t stream) {
    (void)in_sizes; (void)n_in; (void)out_size; (void)ws_size;
    const float* x     = (const float*)d_in[0];
    const float* ln1_g = (const float*)d_in[1];
    const float* ln1_b = (const float*)d_in[2];
    const float* Wqf   = (const float*)d_in[3];
    const float* bqf   = (const float*)d_in[4];
    const float* Wkf   = (const float*)d_in[5];
    const float* bkf   = (const float*)d_in[6];
    const float* Wvf   = (const float*)d_in[7];
    const float* bvf   = (const float*)d_in[8];
    const float* Wof   = (const float*)d_in[9];
    const float* bof   = (const float*)d_in[10];
    const float* ln2_g = (const float*)d_in[11];
    const float* ln2_b = (const float*)d_in[12];
    const float* W1f   = (const float*)d_in[13];
    const float* b1f   = (const float*)d_in[14];
    const float* W2f   = (const float*)d_in[15];
    const float* b2f   = (const float*)d_in[16];
    float* out = (float*)d_out;

    // workspace layout
    char* p = (char*)d_ws;
    size_t cur = 0;
    auto take = [&](size_t bytes) { char* r = p + cur; cur += alignup(bytes); return r; };
    __bf16* Wqkv = (__bf16*)take((size_t)Dq * 3 * Dq * 2);
    float*  bqkv = (float*) take((size_t)3 * Dq * 4);
    __bf16* Wo_b = (__bf16*)take((size_t)Dq * Dq * 2);
    __bf16* W1_b = (__bf16*)take((size_t)Dq * DFFq * 2);
    __bf16* W2_b = (__bf16*)take((size_t)DFFq * Dq * 2);
    __bf16* h_b  = (__bf16*)take((size_t)Mq * Dq * 2);      // reused for h2
    __bf16* qkvb = (__bf16*)take((size_t)Mq * 3 * Dq * 2);
    __bf16* ctxb = (__bf16*)take((size_t)Mq * Dq * 2);
    float*  x1   = (float*) take((size_t)Mq * Dq * 4);
    __bf16* m1b  = (__bf16*)take((size_t)Mq * DFFq * 2);

    // 1. weight prep (bf16)
    {
        int n = 3 * Dq * Dq;
        k_pack_wqkv<<<(n + 255) / 256, 256, 0, stream>>>(Wqf, Wkf, Wvf, Wqkv);
        k_pack_bqkv<<<(3 * Dq + 255) / 256, 256, 0, stream>>>(bqf, bkf, bvf, bqkv);
        k_cvt_bf16<<<(Dq * Dq + 255) / 256, 256, 0, stream>>>(Wof, Wo_b, Dq * Dq);
        k_cvt_bf16<<<(Dq * DFFq + 255) / 256, 256, 0, stream>>>(W1f, W1_b, Dq * DFFq);
        k_cvt_bf16<<<(DFFq * Dq + 255) / 256, 256, 0, stream>>>(W2f, W2_b, DFFq * Dq);
    }

    // 2. LN1
    k_ln<<<Mq, 256, 0, stream>>>(x, ln1_g, ln1_b, h_b);

    // 3. fused QKV projection: [16384,768] x [768,2304]
    k_gemm<1><<<dim3(3 * Dq / 64, Mq / 256), 256, 0, stream>>>(
        h_b, Wqkv, bqkv, nullptr, qkvb, Mq, 3 * Dq, Dq);

    // 4. attention
    k_attn<<<dim3(Sq / 256, Hq, Bq), 256, 0, stream>>>(qkvb, ctxb);

    // 5. output projection + residual(x) -> x1 (f32)
    k_gemm<3><<<dim3(Dq / 64, Mq / 256), 256, 0, stream>>>(
        ctxb, Wo_b, bof, x, x1, Mq, Dq, Dq);

    // 6. LN2 (reuse h_b)
    k_ln<<<Mq, 256, 0, stream>>>(x1, ln2_g, ln2_b, h_b);

    // 7. MLP up + exact GELU: [16384,768] x [768,3072]
    k_gemm<2><<<dim3(DFFq / 64, Mq / 256), 256, 0, stream>>>(
        h_b, W1_b, b1f, nullptr, m1b, Mq, DFFq, Dq);

    // 8. MLP down + residual(x1) -> out (f32): [16384,3072] x [3072,768]
    k_gemm<3><<<dim3(Dq / 64, Mq / 256), 256, 0, stream>>>(
        m1b, W2_b, b2f, x1, out, Mq, Dq, DFFq);
}
